// LSTMhead_68255620268784
// MI455X (gfx1250) — compile-verified, matrix operand order checked by reading, not run on hardware
//
#include <hip/hip_runtime.h>
#include <math.h>

typedef __attribute__((ext_vector_type(16))) __bf16 bf16x16;
typedef __attribute__((ext_vector_type(8)))  float  f32x8;

union Frag { bf16x16 v; unsigned short s[16]; };

__device__ __forceinline__ unsigned short f2bf(float f) {
    union { float f; unsigned u; } v; v.f = f;
    unsigned u = v.u;
    unsigned r = u + 0x7FFFu + ((u >> 16) & 1u);   // round-to-nearest-even
    return (unsigned short)(r >> 16);
}

// ---------------------------------------------------------------------------
// Kernel 1: fused 3x3 conv (+b1, ReLU) -> 1x1 conv (+b2), via bf16 WMMA.
// Block = 256 threads (8 waves). Block tile: M=64 channels x N=32 spatial
// columns of one row h of one image b. Waves: mt = wid&3 (M tile), nt = wid>>2.
// GEMM1: K = 64*9 = 576 over 18 chunks of 32.  GEMM2: K = 64 over 2 chunks.
// Both WMMA operands are fed from K-major LDS panels so all fragment loads
// are contiguous 16B (ds_load_b128) per 8-element group.
// ---------------------------------------------------------------------------
__global__ __launch_bounds__(256)
void conv_fused_kernel(const float* __restrict__ x,
                       const float* __restrict__ w1, const float* __restrict__ b1,
                       const float* __restrict__ w2, const float* __restrict__ b2,
                       float* __restrict__ fout)
{
    __shared__ unsigned short sX[64 * 3 * 36];   // halo patch, bf16, col pad 36
    __shared__ unsigned short sA[64 * 32];       // per-chunk w1 panel [c1][kk]
    __shared__ unsigned short sB[32 * 32];       // per-chunk im2col panel [n][kk]
    __shared__ unsigned short sY1T[32 * 64];     // conv1 out, transposed [n][c1]
    __shared__ unsigned short sW2[64 * 64];      // 1x1 weights [d][c1]
    __shared__ float sB1[64], sB2[64];

    const int tid  = threadIdx.x;
    const int lane = tid & 31;
    const int half = lane >> 4;          // K-group selector (wave32)
    const int ln16 = lane & 15;          // M (for A) / N (for B,C,D)
    const int wid  = tid >> 5;
    const int mt   = wid & 3;            // M tile: channels mt*16..mt*16+15
    const int nt   = wid >> 2;           // N tile: cols  nt*16..nt*16+15

    const int w0 = blockIdx.x * 32;
    const int h  = blockIdx.y;
    const int b  = blockIdx.z;

    // Stage halo patch: channels x 3 rows x 34 cols (zero padded), as bf16.
    for (int idx = tid; idx < 64 * 3 * 34; idx += 256) {
        int c0  = idx / (3 * 34);
        int rem = idx % (3 * 34);
        int r   = rem / 34;
        int col = rem % 34;
        int hh  = h - 1 + r;
        int ww  = w0 - 1 + col;
        float v = 0.0f;
        if (hh >= 0 && hh < 128 && ww >= 0 && ww < 128)
            v = x[(((size_t)b * 64 + c0) * 128 + hh) * 128 + ww];
        sX[(c0 * 3 + r) * 36 + col] = f2bf(v);
    }
    // Stage 1x1 weights and biases.
    for (int idx = tid; idx < 64 * 64; idx += 256)
        sW2[idx] = f2bf(w2[idx]);
    if (tid < 64) { sB1[tid] = b1[tid]; sB2[tid] = b2[tid]; }

    // ---------------- GEMM1: y1 = relu(w1 (x) patch + b1) ----------------
    f32x8 acc = {};
    for (int kc = 0; kc < 18; ++kc) {
        __syncthreads();
        // Stage w1 K-chunk: sA[c1][kk], k = kc*32+kk, k = c0*9 + (kh*3 + kw).
        for (int idx = tid; idx < 64 * 32; idx += 256) {
            int c1 = idx >> 5, kk = idx & 31;
            int k  = kc * 32 + kk;
            int c0 = k / 9, t = k % 9;
            sA[c1 * 32 + kk] = f2bf(w1[(c1 * 64 + c0) * 9 + t]);
        }
        // Stage im2col K-chunk: sB[n][kk] gathered once per block from sX.
        for (int idx = tid; idx < 32 * 32; idx += 256) {
            int n  = idx >> 5, kk = idx & 31;
            int k  = kc * 32 + kk;
            int c0 = k / 9, t = k % 9, kh = t / 3, kw = t % 3;
            sB[n * 32 + kk] = sX[(c0 * 3 + kh) * 36 + (n + kw)];
        }
        __syncthreads();

        Frag a, bb;
        const unsigned short* aRow = &sA[(mt * 16 + ln16) * 32];
        const unsigned short* bRow = &sB[(nt * 16 + ln16) * 32];
        #pragma unroll
        for (int e = 0; e < 16; ++e) {
            int kl = ((e >> 3) << 4) + half * 8 + (e & 7);   // ISA 16-bit layout
            a.s[e]  = aRow[kl];
            bb.s[e] = bRow[kl];
        }
        acc = __builtin_amdgcn_wmma_f32_16x16x32_bf16(
                  false, a.v, false, bb.v, (short)0, acc, false, false);
    }

    // Bias + ReLU; restage transposed [n][c1] so GEMM2 B-loads are contiguous.
    {
        int n = nt * 16 + ln16;
        #pragma unroll
        for (int r = 0; r < 8; ++r) {
            int m = mt * 16 + half * 8 + r;       // C/D layout: VGPR r -> M=r / r+8
            float y = acc[r] + sB1[m];
            y = y > 0.0f ? y : 0.0f;
            sY1T[n * 64 + m] = f2bf(y);           // contiguous over r
        }
    }
    __syncthreads();

    // ---------------- GEMM2: f = w2 * y1 + b2 ----------------
    f32x8 acc2 = {};
    {
        const unsigned short* aRow = &sW2[(mt * 16 + ln16) * 64];
        const unsigned short* bRow = &sY1T[(nt * 16 + ln16) * 64];
        #pragma unroll
        for (int kc = 0; kc < 2; ++kc) {
            Frag a, bb;
            #pragma unroll
            for (int e = 0; e < 16; ++e) {
                int kl = ((e >> 3) << 4) + half * 8 + (e & 7);
                int c1 = kc * 32 + kl;
                a.s[e]  = aRow[c1];
                bb.s[e] = bRow[c1];
            }
            acc2 = __builtin_amdgcn_wmma_f32_16x16x32_bf16(
                       false, a.v, false, bb.v, (short)0, acc2, false, false);
        }
    }

    #pragma unroll
    for (int r = 0; r < 8; ++r) {
        int d = mt * 16 + half * 8 + r;
        int n = nt * 16 + ln16;
        fout[((size_t)b * 64 + d) * 16384 + h * 128 + (w0 + n)] = acc2[r] + sB2[d];
    }
}

// ---------------------------------------------------------------------------
// Kernel 2: exact top-128 (sorted descending) per row of 16384 via 4-pass
// MSB radix select + collect + 128-wide bitonic sort in LDS.
// One block (256 threads) per row; 1024 rows.
// ---------------------------------------------------------------------------
__device__ __forceinline__ unsigned fkey(float f) {
    union { float f; unsigned u; } v; v.f = f;
    unsigned u = v.u;
    return (u & 0x80000000u) ? ~u : (u | 0x80000000u);   // monotone, descending-safe
}

__global__ __launch_bounds__(256)
void topk128_kernel(const float* __restrict__ fsrc, float* __restrict__ vals)
{
    __shared__ unsigned hist[256];
    __shared__ unsigned s_prefix, s_maskHigh, s_remaining;
    __shared__ float    svals[128];
    __shared__ unsigned cntG, cntE;

    const int tid = threadIdx.x;
    const float* row = fsrc + (size_t)blockIdx.x * 16384;

    if (tid == 0) { s_prefix = 0u; s_maskHigh = 0u; s_remaining = 128u; }
    __syncthreads();

    for (int bp = 3; bp >= 0; --bp) {
        hist[tid] = 0u;
        __syncthreads();
        unsigned maskHigh = s_maskHigh, prefix = s_prefix;
        for (int i = tid; i < 16384; i += 256) {
            unsigned k = fkey(row[i]);
            if ((k & maskHigh) == prefix)
                atomicAdd(&hist[(k >> (bp * 8)) & 0xFF], 1u);
        }
        __syncthreads();
        if (tid == 0) {
            unsigned rem = s_remaining, cum = 0u;
            int v;
            for (v = 255; v > 0; --v) {
                if (cum + hist[v] >= rem) break;
                cum += hist[v];
            }
            s_prefix    = prefix   | ((unsigned)v << (bp * 8));
            s_maskHigh  = maskHigh | (0xFFu       << (bp * 8));
            s_remaining = rem - cum;
        }
        __syncthreads();
    }

    const unsigned T   = s_prefix;
    const unsigned rem = s_remaining;        // # of threshold-equal elems to keep
    const unsigned nG  = 128u - rem;         // # strictly greater than threshold
    if (tid == 0) { cntG = 0u; cntE = 0u; }
    __syncthreads();

    for (int i = tid; i < 16384; i += 256) {
        float x = row[i];
        unsigned k = fkey(x);
        if (k > T) {
            unsigned p = atomicAdd(&cntG, 1u);
            svals[p] = x;                    // p < nG by construction
        } else if (k == T) {
            unsigned p = atomicAdd(&cntE, 1u);
            if (p < rem) svals[nG + p] = x;
        }
    }
    __syncthreads();

    // Bitonic sort, 128 elems, descending.
    for (unsigned k = 2; k <= 128; k <<= 1) {
        for (unsigned j = k >> 1; j > 0; j >>= 1) {
            __syncthreads();
            if (tid < 128) {
                unsigned ixj = tid ^ j;
                if (ixj > tid) {
                    float a = svals[tid], b = svals[ixj];
                    bool desc = ((tid & k) == 0);
                    if (desc ? (a < b) : (a > b)) { svals[tid] = b; svals[ixj] = a; }
                }
            }
        }
    }
    __syncthreads();
    if (tid < 128)
        vals[(size_t)blockIdx.x * 128 + tid] = svals[tid];
}

// ---------------------------------------------------------------------------
// Kernel 3: tiny LSTM, hid=2, seq=128, batch=16. One persistent block;
// 128 threads = 16 batches x 8 gates. State lives in LDS.
// ---------------------------------------------------------------------------
__device__ __forceinline__ float sigmoidf(float x) { return 1.0f / (1.0f + __expf(-x)); }

__global__ __launch_bounds__(128)
void lstm_kernel(const float* __restrict__ vals,       // [16][64][128]
                 const float* __restrict__ w_ih,       // [8][64]
                 const float* __restrict__ w_hh,       // [8][2]
                 const float* __restrict__ b_ih, const float* __restrict__ b_hh,
                 float* __restrict__ out)              // [16][128][2]
{
    __shared__ float sWih[8 * 64], sWhh[16], sB[8];
    __shared__ float hS[32], cS[32], gS[128];

    const int tid = threadIdx.x;
    for (int i = tid; i < 8 * 64; i += 128) sWih[i] = w_ih[i];
    if (tid < 16) sWhh[tid] = w_hh[tid];
    if (tid < 8)  sB[tid]   = b_ih[tid] + b_hh[tid];
    if (tid < 32) { hS[tid] = 0.0f; cS[tid] = 0.0f; }
    __syncthreads();

    const int b = tid >> 3;
    const int g = tid & 7;
    const float* xr = vals + (size_t)b * 64 * 128;

    for (int t = 0; t < 128; ++t) {
        float s = sB[g];
        #pragma unroll 8
        for (int d = 0; d < 64; ++d)
            s += xr[d * 128 + t] * sWih[g * 64 + d];
        s += hS[b * 2 + 0] * sWhh[g * 2 + 0] + hS[b * 2 + 1] * sWhh[g * 2 + 1];
        gS[tid] = s;
        __syncthreads();

        if (tid < 32) {
            int bb = tid >> 1, hh = tid & 1;
            float ig = sigmoidf(gS[bb * 8 + 0 + hh]);
            float fg = sigmoidf(gS[bb * 8 + 2 + hh]);
            float gg = tanhf   (gS[bb * 8 + 4 + hh]);
            float og = sigmoidf(gS[bb * 8 + 6 + hh]);
            float c  = fg * cS[tid] + ig * gg;
            cS[tid]  = c;
            float hv = og * tanhf(c);
            hS[tid]  = hv;
            out[(bb * 128 + t) * 2 + hh] = hv;
        }
        __syncthreads();
    }
}

// ---------------------------------------------------------------------------
extern "C" void kernel_launch(void* const* d_in, const int* in_sizes, int n_in,
                              void* d_out, int out_size, void* d_ws, size_t ws_size,
                              hipStream_t stream)
{
    const float* x    = (const float*)d_in[0];
    const float* w1   = (const float*)d_in[1];
    const float* b1   = (const float*)d_in[2];
    const float* w2   = (const float*)d_in[3];
    const float* b2   = (const float*)d_in[4];
    const float* w_ih = (const float*)d_in[5];
    const float* w_hh = (const float*)d_in[6];
    const float* b_ih = (const float*)d_in[7];
    const float* b_hh = (const float*)d_in[8];

    float* f    = (float*)d_ws;                        // [16][64][16384]
    float* vals = f + (size_t)16 * 64 * 16384;         // [16][64][128]

    conv_fused_kernel<<<dim3(4, 128, 16), 256, 0, stream>>>(x, w1, b1, w2, b2, f);
    topk128_kernel<<<dim3(16 * 64), 256, 0, stream>>>(f, vals);
    lstm_kernel<<<dim3(1), 128, 0, stream>>>(vals, w_ih, w_hh, b_ih, b_hh, (float*)d_out);
}
